// MultiHeadAttention_78056735637728
// MI455X (gfx1250) — matrix operand order repaired and checked by reading, not compiled, not run on hardware
//
#include <hip/hip_runtime.h>

typedef __attribute__((ext_vector_type(16))) _Float16 v16h;
typedef __attribute__((ext_vector_type(8)))  _Float16 v8h;
typedef __attribute__((ext_vector_type(8)))  float    v8f;
typedef __attribute__((ext_vector_type(4)))  float    v4f;
typedef __attribute__((ext_vector_type(4)))  int      v4i;

#define D_MODEL 256
#define N_HEADS 8
#define HEAD_DIM 32
#define BATCH 8
#define SEQ 2048
#define ROWS (BATCH * SEQ)   // 16384

#if defined(__has_builtin)
#if __has_builtin(__builtin_amdgcn_global_load_async_to_lds_b128)
#define USE_ASYNC 1
#endif
#endif

#if defined(USE_ASYNC)
// Builtin signature (from hipcc diagnostic): (AS1 v4i*, AS3 v4i*, Imm int, Imm int)
#define ASYNC_B128(g, l, off) \
    __builtin_amdgcn_global_load_async_to_lds_b128( \
        (__attribute__((address_space(1))) v4i*)(g), \
        (__attribute__((address_space(3))) v4i*)(l), (off), 0)
#endif

// ---------------------------------------------------------------------------
// Weight transpose + f32->f16 convert:  Wt[n][k] = (f16) W[k][n]
// ---------------------------------------------------------------------------
__global__ __launch_bounds__(256)
void wT_kernel(const float* __restrict__ W, _Float16* __restrict__ Wt) {
    int idx = blockIdx.x * 256 + threadIdx.x;   // 0 .. 65535
    int n = idx >> 8;
    int k = idx & 255;
    Wt[n * 256 + k] = (_Float16)W[k * 256 + n];
}

// ---------------------------------------------------------------------------
// Projection: Y = X(f32) @ W + b, emitted as f16 in head-major layouts.
//   vmode==0 : out[((b*H + h)*S + s)*32 + d]      (Q, K)
//   vmode==1 : out[((b*H + h)*32 + d)*S + s]      (V, transposed for P*V B-frags)
// ---------------------------------------------------------------------------
__global__ __launch_bounds__(32)
void proj_kernel(const float* __restrict__ X, const _Float16* __restrict__ Wt,
                 const float* __restrict__ bias, _Float16* __restrict__ out,
                 int vmode) {
    const int lane = threadIdx.x;
    const int i0 = blockIdx.x * 16;     // output row tile (b*S+s space)
    const int n0 = blockIdx.y * 16;     // output feature tile
    const int lm   = lane & 15;
    const int roff = (lane < 16) ? 0 : 8;   // C/D row offset
    const int kb   = (lane < 16) ? 0 : 16;  // B k-base

    v8f acc = {};
    for (int k0 = 0; k0 < 256; k0 += 32) {
        // A fragment: row i0+lm of X, k = k0 + kb + e  (contiguous)
        const float* arow = X + (size_t)(i0 + lm) * 256 + k0 + kb;
        v4f a0 = *(const v4f*)(arow + 0);
        v4f a1 = *(const v4f*)(arow + 4);
        v4f a2 = *(const v4f*)(arow + 8);
        v4f a3 = *(const v4f*)(arow + 12);
        v16h a;
#pragma unroll
        for (int e = 0; e < 4; ++e) {
            a[e]      = (_Float16)a0[e];
            a[4 + e]  = (_Float16)a1[e];
            a[8 + e]  = (_Float16)a2[e];
            a[12 + e] = (_Float16)a3[e];
        }
        // B fragment: Wt row (n0+lm), k = k0 + kb + e  (contiguous 32B)
        v16h b = *(const v16h*)(Wt + (size_t)(n0 + lm) * 256 + k0 + kb);
        acc = __builtin_amdgcn_wmma_f32_16x16x32_f16(false, a, false, b,
                                                     (short)0, acc, false, false);
    }
    const float bval = bias[n0 + lm];
#pragma unroll
    for (int r = 0; r < 8; ++r) {
        int row = i0 + roff + r;              // global row = b*S + s
        int col = n0 + lm;                    // feature in [0,256)
        float y = acc[r] + bval;
        int bb = row >> 11, s = row & 2047;
        int h = col >> 5, d = col & 31;
        size_t off;
        if (vmode)
            off = ((size_t)(bb * N_HEADS + h) * HEAD_DIM + d) * SEQ + s;
        else
            off = ((size_t)(bb * N_HEADS + h) * SEQ + s) * HEAD_DIM + d;
        out[off] = (_Float16)y;
    }
}

// ---------------------------------------------------------------------------
// Flash-style attention. One wave per (b, h, 16 query rows).
// 32 keys/step: 2 QK^T WMMAs, LDS round-trip for softmax + P re-layout,
// 2 P*V WMMAs. K/V tiles double-buffered in LDS via async global->LDS copies
// (ASYNCcnt) when the toolchain exposes them.
// ---------------------------------------------------------------------------
__global__ __launch_bounds__(32)
void attn_kernel(const _Float16* __restrict__ Qh, const _Float16* __restrict__ Kh,
                 const _Float16* __restrict__ VhT, const float* __restrict__ mask,
                 _Float16* __restrict__ Ch) {
    __shared__ float    mb[SEQ];        // mask * -1e9
    __shared__ float    slds[16 * 32];  // logits tile
    __shared__ __align__(64) _Float16 plds[16 * 32];  // probs tile
    __shared__ float    sf[16];         // per-row rescale factors
    __shared__ float    rl[16];         // per-row 1/sum
#if defined(USE_ASYNC)
    __shared__ __align__(64) _Float16 ktile[2][32 * 32];  // [j'][d]
    __shared__ __align__(64) _Float16 vtile[2][32 * 32];  // [d][j']
#endif

    const int lane = threadIdx.x;
    const int bid  = blockIdx.x;            // 0 .. B*H*(S/16)-1
    const int qt   = bid & 127;
    const int bh   = bid >> 7;              // b*H + h
    const int b    = bh >> 3;
    const int h    = bh & 7;
    const int i0   = qt * 16;

    for (int j = lane; j < SEQ; j += 32) mb[j] = mask[j] * (-1e9f);

    const int lm   = lane & 15;
    const int roff = (lane < 16) ? 0 : 8;   // C row offset
    const int kb   = (lane < 16) ? 0 : 16;  // B k-base

    const _Float16* Kbase = Kh  + (size_t)bh * SEQ * HEAD_DIM;
    const _Float16* Vbase = VhT + (size_t)bh * HEAD_DIM * SEQ;

#if defined(USE_ASYNC)
    // Issue async copy of one 32x32 K tile (contiguous 2KB) and one 32x32 V
    // tile (32 strided 64B rows) into LDS buffer `buf`. 8 x b128 per lane set.
    auto issue_copy = [&](int buf, int j0) {
        // K: tile is contiguous; instruction imm offset applies to both
        // global and LDS address -> one base, 4 chunks of 512B.
        const _Float16* gk = Kbase + (size_t)j0 * HEAD_DIM + lane * 8;
        _Float16* lk = &ktile[buf][lane * 8];
        ASYNC_B128(gk, lk, 0);
        ASYNC_B128(gk, lk, 512);
        ASYNC_B128(gk, lk, 1024);
        ASYNC_B128(gk, lk, 1536);
        // V: global row stride is SEQ halfs, LDS row stride 32 halfs -> per-
        // chunk addresses. 4 lanes cover one 64B row.
        const int vrow = lane >> 2;
        const int vseg = (lane & 3) * 8;
#pragma unroll
        for (int c = 0; c < 4; ++c) {
            const _Float16* gv = Vbase + (size_t)(c * 8 + vrow) * SEQ + j0 + vseg;
            _Float16* lv = &vtile[buf][(c * 8 + vrow) * 32 + vseg];
            ASYNC_B128(gv, lv, 0);
        }
    };
#endif

    // Q A-fragment (head dim is exactly K=32): row i0+lm
    const _Float16* qrow = Qh + ((size_t)bh * SEQ + i0 + lm) * HEAD_DIM;
    v16h qa;
    {
        v8h c0 = *(const v8h*)(qrow + kb);          // k = kb .. kb+7
        v8h c1 = *(const v8h*)(qrow + kb + 8);      // k = kb+8 .. kb+15
#pragma unroll
        for (int e = 0; e < 8; ++e) { qa[e] = c0[e]; qa[8 + e] = c1[e]; }
    }

#if defined(USE_ASYNC)
    issue_copy(0, 0);
#endif
    __syncthreads();

    float row_m = -3.0e38f, row_l = 0.0f;
    v8f o0 = {}, o1 = {};
    const float inv_s = 0.17677669529663687f;   // 1/sqrt(32)
    const int cb = (lane < 16) ? 0 : 16;        // softmax column half

    for (int jt = 0; jt < SEQ / 32; ++jt) {
        const int j0 = jt * 32;
#if defined(USE_ASYNC)
        const int buf = jt & 1;
        if (jt + 1 < SEQ / 32) {
            issue_copy(buf ^ 1, j0 + 32);                 // prefetch next tile
            asm volatile("s_wait_asynccnt 0x8" ::: "memory");  // current done
        } else {
            asm volatile("s_wait_asynccnt 0x0" ::: "memory");
        }
        v16h kf0 = *(const v16h*)&ktile[buf][lm * 32 + kb];
        v16h kf1 = *(const v16h*)&ktile[buf][(16 + lm) * 32 + kb];
#else
        v16h kf0 = *(const v16h*)(Kbase + (size_t)(j0 + lm) * HEAD_DIM + kb);
        v16h kf1 = *(const v16h*)(Kbase + (size_t)(j0 + 16 + lm) * HEAD_DIM + kb);
#endif
        v8f s0 = {}, s1 = {};
        s0 = __builtin_amdgcn_wmma_f32_16x16x32_f16(false, qa, false, kf0,
                                                    (short)0, s0, false, false);
        s1 = __builtin_amdgcn_wmma_f32_16x16x32_f16(false, qa, false, kf1,
                                                    (short)0, s1, false, false);
        // Spill logits tile to LDS (C-frag layout -> plain [row][col])
#pragma unroll
        for (int r = 0; r < 8; ++r) {
            slds[(roff + r) * 32 + lm]      = s0[r];
            slds[(roff + r) * 32 + 16 + lm] = s1[r];
        }
        __syncthreads();

        // Online softmax, 2 lanes per row (16 cols each); lane pair (l, l^16)
        // combines partial max/sum via ds_swizzle SWAPX16.
        {
            float vv[16];
            float pmx = -3.0e38f;
#pragma unroll
            for (int c = 0; c < 16; ++c) {
                vv[c] = slds[lm * 32 + cb + c] * inv_s + mb[j0 + cb + c];
                pmx = fmaxf(pmx, vv[c]);
            }
            float omx = __int_as_float(
                __builtin_amdgcn_ds_swizzle(__float_as_int(pmx), 0x401F));
            float mx = fmaxf(fmaxf(pmx, omx), row_m);
            float corr = __expf(row_m - mx);
            float psum = 0.0f;
#pragma unroll
            for (int c = 0; c < 16; ++c) {
                float p = __expf(vv[c] - mx);
                psum += p;
                plds[lm * 32 + cb + c] = (_Float16)p;
            }
            float osum = __int_as_float(
                __builtin_amdgcn_ds_swizzle(__float_as_int(psum), 0x401F));
            row_l = row_l * corr + (psum + osum);
            row_m = mx;
            if (lane < 16) sf[lm] = corr;
        }
        __syncthreads();

        // Rescale O accumulators by per-row factor (broadcast via LDS)
        {
            v4f c0 = *(const v4f*)(sf + roff);
            v4f c1 = *(const v4f*)(sf + roff + 4);
#pragma unroll
            for (int r = 0; r < 4; ++r) {
                o0[r] *= c0[r];     o1[r] *= c0[r];
                o0[4 + r] *= c1[r]; o1[4 + r] *= c1[r];
            }
        }
        // P A-fragment from LDS: row lm, k = kb + e
        v16h pa;
        {
            const _Float16* pr = plds + lm * 32 + kb;
            v8h c0 = *(const v8h*)(pr);
            v8h c1 = *(const v8h*)(pr + 8);
#pragma unroll
            for (int e = 0; e < 8; ++e) { pa[e] = c0[e]; pa[8 + e] = c1[e]; }
        }
        // V B-fragments: B[k=j'][n=d] = VhT[d][j0+k] -> contiguous in k
#if defined(USE_ASYNC)
        v16h vf0 = *(const v16h*)&vtile[buf][lm * 32 + kb];
        v16h vf1 = *(const v16h*)&vtile[buf][(16 + lm) * 32 + kb];
#else
        v16h vf0 = *(const v16h*)(Vbase + (size_t)lm * SEQ + j0 + kb);
        v16h vf1 = *(const v16h*)(Vbase + (size_t)(16 + lm) * SEQ + j0 + kb);
#endif
        o0 = __builtin_amdgcn_wmma_f32_16x16x32_f16(false, pa, false, vf0,
                                                    (short)0, o0, false, false);
        o1 = __builtin_amdgcn_wmma_f32_16x16x32_f16(false, pa, false, vf1,
                                                    (short)0, o1, false, false);
        __syncthreads();   // protect slds/plds before next step overwrites
    }

    if (lane < 16) rl[lm] = 1.0f / row_l;
    __syncthreads();
    {
        v4f c0 = *(const v4f*)(rl + roff);
        v4f c1 = *(const v4f*)(rl + roff + 4);
#pragma unroll
        for (int r = 0; r < 4; ++r) {
            o0[r] *= c0[r];     o1[r] *= c0[r];
            o0[4 + r] *= c1[r]; o1[4 + r] *= c1[r];
        }
    }
    // Store context: Ch[b*S+s][h*32 + d], f16
#pragma unroll
    for (int r = 0; r < 8; ++r) {
        int s = i0 + roff + r;
        _Float16* crow = Ch + ((size_t)(b * SEQ + s)) * D_MODEL + h * HEAD_DIM;
        crow[lm]      = (_Float16)o0[r];
        crow[16 + lm] = (_Float16)o1[r];
    }
}

// ---------------------------------------------------------------------------
// Output projection: out(f32) = Ch(f16) @ wo + bo
// ---------------------------------------------------------------------------
__global__ __launch_bounds__(32)
void oproj_kernel(const _Float16* __restrict__ Ch, const _Float16* __restrict__ WoT,
                  const float* __restrict__ bias, float* __restrict__ out) {
    const int lane = threadIdx.x;
    const int i0 = blockIdx.x * 16;
    const int n0 = blockIdx.y * 16;
    const int lm   = lane & 15;
    const int roff = (lane < 16) ? 0 : 8;
    const int kb   = (lane < 16) ? 0 : 16;

    v8f acc = {};
    for (int k0 = 0; k0 < 256; k0 += 32) {
        const _Float16* arow = Ch + (size_t)(i0 + lm) * 256 + k0 + kb;
        v8h c0 = *(const v8h*)(arow);
        v8h c1 = *(const v8h*)(arow + 8);
        v16h a;
#pragma unroll
        for (int e = 0; e < 8; ++e) { a[e] = c0[e]; a[8 + e] = c1[e]; }
        v16h bfrag = *(const v16h*)(WoT + (size_t)(n0 + lm) * 256 + k0 + kb);
        acc = __builtin_amdgcn_wmma_f32_16x16x32_f16(false, a, false, bfrag,
                                                     (short)0, acc, false, false);
    }
    const float bval = bias[n0 + lm];
#pragma unroll
    for (int r = 0; r < 8; ++r) {
        out[(size_t)(i0 + roff + r) * 256 + n0 + lm] = acc[r] + bval;
    }
}

// ---------------------------------------------------------------------------
extern "C" void kernel_launch(void* const* d_in, const int* in_sizes, int n_in,
                              void* d_out, int out_size, void* d_ws, size_t ws_size,
                              hipStream_t stream) {
    const float* q  = (const float*)d_in[0];
    const float* k  = (const float*)d_in[1];
    const float* v  = (const float*)d_in[2];
    const float* m  = (const float*)d_in[3];
    const float* wq = (const float*)d_in[4];
    const float* bq = (const float*)d_in[5];
    const float* wk = (const float*)d_in[6];
    const float* bk = (const float*)d_in[7];
    const float* wv = (const float*)d_in[8];
    const float* bv = (const float*)d_in[9];
    const float* wo = (const float*)d_in[10];
    const float* bo = (const float*)d_in[11];
    float* out = (float*)d_out;

    _Float16* ws = (_Float16*)d_ws;
    const size_t WE = 256 * 256;                                // per weight
    const size_t HE = (size_t)BATCH * N_HEADS * SEQ * HEAD_DIM; // 4,194,304
    _Float16* wqT = ws;
    _Float16* wkT = wqT + WE;
    _Float16* wvT = wkT + WE;
    _Float16* woT = wvT + WE;
    _Float16* Qh  = woT + WE;
    _Float16* Kh  = Qh + HE;
    _Float16* VhT = Kh + HE;
    _Float16* Ch  = VhT + HE;

    wT_kernel<<<256, 256, 0, stream>>>(wq, wqT);
    wT_kernel<<<256, 256, 0, stream>>>(wk, wkT);
    wT_kernel<<<256, 256, 0, stream>>>(wv, wvT);
    wT_kernel<<<256, 256, 0, stream>>>(wo, woT);

    dim3 pgrid(ROWS / 16, D_MODEL / 16);   // 1024 x 16 tiles
    proj_kernel<<<pgrid, 32, 0, stream>>>(q, wqT, bq, Qh, 0);
    proj_kernel<<<pgrid, 32, 0, stream>>>(k, wkT, bk, Kh, 0);
    proj_kernel<<<pgrid, 32, 0, stream>>>(v, wvT, bv, VhT, 1);

    attn_kernel<<<BATCH * N_HEADS * (SEQ / 16), 32, 0, stream>>>(Qh, Kh, VhT, m, Ch);

    oproj_kernel<<<pgrid, 32, 0, stream>>>(Ch, woT, bo, out);
}
